// ModifiedSelectiveAttentionModule_19430432047675
// MI455X (gfx1250) — matrix operand order repaired and checked by reading, not compiled, not run on hardware
//
#include <hip/hip_runtime.h>
#include <cstdint>
#include <cstddef>

#define B_  8
#define S_  2048
#define E_  1024
#define DK_ 512
#define DV_ 512

typedef __attribute__((ext_vector_type(16))) __bf16 v16bf;
typedef __attribute__((ext_vector_type(8)))  __bf16 v8bf;
typedef __attribute__((ext_vector_type(8)))  float  v8f;
typedef __attribute__((ext_vector_type(4)))  unsigned int u32x4;
typedef __attribute__((ext_vector_type(4)))  int i32x4;
typedef __attribute__((ext_vector_type(8)))  int i32x8;

#ifndef __has_builtin
#define __has_builtin(x) 0
#endif
#if __has_builtin(__builtin_amdgcn_tensor_load_to_lds)
#define HAVE_TDM 1
#else
#define HAVE_TDM 0
#endif

static __device__ __forceinline__ __bf16 f2bf(float f) {
    union { float f; uint32_t u; } v; v.f = f;
    uint32_t r = v.u + 0x7FFFu + ((v.u >> 16) & 1u);
    uint16_t h = (uint16_t)(r >> 16);
    __bf16 o; __builtin_memcpy(&o, &h, 2); return o;
}

static __device__ __forceinline__ float sigmoidf_(float x) {
    return 1.0f / (1.0f + __expf(-x));
}
static __device__ __forceinline__ float geluf_(float x) {
    return 0.5f * x * (1.0f + erff(x * 0.70710678118f));
}

// Workgroup-relative LDS byte offset: generic LDS address is {aperture, off[31:0]}.
static __device__ __forceinline__ unsigned lds_addr_of(const void* p) {
    return (unsigned)(unsigned long long)p;
}

#if HAVE_TDM
// TDM 2D tile load: rows x cols elements of 2-byte data, row stride in elements.
// D# group0: count=1 | lds_addr | global_addr[56:0] | type=2.
// D# group1: data_size=1(2B); tensor_dim0=cols; tensor_dim1=rows; tile_dim0=cols;
//            tile_dim1=rows; tensor_dim0_stride=rstride.  (ISA 8.3/8.4)
static __device__ __forceinline__ void tdm_load_2d(const void* gsrc, unsigned lds_off,
                                                   int rows, int cols, int rstride) {
    unsigned long long ga = (unsigned long long)gsrc;
    u32x4 g0;
    g0[0] = 1u;                                                     // count=1
    g0[1] = lds_off;                                                // lds_addr
    g0[2] = (unsigned)ga;                                           // global_addr lo
    g0[3] = (unsigned)((ga >> 32) & 0x01FFFFFFull) | (2u << 30);    // addr hi | type=2
    i32x8 g1;
    g1[0] = 1 << 16;                                                // data_size = 2B
    g1[1] = (cols & 0xFFFF) << 16;                                  // tensor_dim0 lo16
    g1[2] = ((cols >> 16) & 0xFFFF) | ((rows & 0xFFFF) << 16);      // dim0 hi | dim1 lo
    g1[3] = ((rows >> 16) & 0xFFFF) | ((cols & 0xFFFF) << 16);      // dim1 hi | tile_dim0
    g1[4] = rows & 0xFFFF;                                          // tile_dim1
    g1[5] = rstride;                                                // dim0 stride lo32
    g1[6] = 0;
    g1[7] = 0;
    i32x4 z4 = {};
#if defined(__clang_major__) && (__clang_major__ >= 23)
    i32x8 z8 = {};
    __builtin_amdgcn_tensor_load_to_lds(g0, g1, z4, z4, z8, 0);
#else
    __builtin_amdgcn_tensor_load_to_lds(g0, g1, z4, z4, 0);
#endif
}
#endif

// A fragment: 16x32 bf16 (M x K). lane&15 = row M; elems 0-15 = K[k0+ko .. +15],
// ko = (lane>=16)*16.  (contiguous k order, same as load_b_frag)
static __device__ __forceinline__ v16bf load_a_frag(const __bf16* A, int lda, int k0, int lane) {
    int row = lane & 15;
    int ko  = k0 + ((lane >> 4) << 4);
    const __bf16* p = A + (size_t)row * lda + ko;
    v8bf lo = *(const v8bf*)p;
    v8bf hi = *(const v8bf*)(p + 8);
    v16bf r;
#pragma unroll
    for (int i = 0; i < 8; ++i) { r[i] = lo[i]; r[i + 8] = hi[i]; }
    return r;
}

// B fragment: 32x16 bf16 (K x N), source stored row-major [N, K].
// lane&15 = col N; 16 contiguous K at k0 + (lane>=16)*16.
static __device__ __forceinline__ v16bf load_b_frag(const __bf16* Bm, int ldb, int k0, int lane) {
    int n  = lane & 15;
    int ko = k0 + ((lane >> 4) << 4);
    return *(const v16bf*)(Bm + (size_t)n * ldb + ko);
}

// B fragment from a K-major source: element (k,n) = V[k*ldv + n] (for P@V).
static __device__ __forceinline__ v16bf load_b_frag_kmajor(const __bf16* V, int ldv, int nbase, int lane) {
    int n  = nbase + (lane & 15);
    int ko = (lane >> 4) << 4;
    const __bf16* p = V + (size_t)ko * ldv + n;
    v16bf r;
#pragma unroll
    for (int i = 0; i < 16; ++i) r[i] = p[(size_t)i * ldv];
    return r;
}

// ---------------------------------------------------------------- convert
__global__ void k_f32_to_bf16(const float* __restrict__ src, __bf16* __restrict__ dst, int n) {
    int i = blockIdx.x * blockDim.x + threadIdx.x;
    if (i < n) dst[i] = f2bf(src[i]);
}

// ---------------------------------------------------------------- GEMM
// C[M,N] = A[M,K](bf16,row-major) @ W[N,K]^T (bf16,row-major) + bias
// A strip (16 x K) staged once per block into LDS via TDM; 8 waves share it.
// grid = (N/128, M/16), block = 256 (8 waves); wave w -> N tile w.
__global__ void k_gemm_bf16(const __bf16* __restrict__ A, const __bf16* __restrict__ W,
                            const float* __restrict__ bias,
                            float* __restrict__ Cf, __bf16* __restrict__ Cb, __bf16* __restrict__ Cg,
                            int M, int N, int K) {
    __shared__ __align__(128) __bf16 ashm[16 * E_];   // up to 32 KB

    int lane  = threadIdx.x & 31;
    int wave  = threadIdx.x >> 5;
    int nbase = blockIdx.x * 128 + wave * 16;
    int mbase = blockIdx.y * 16;
    (void)M;

    const __bf16* Ab = A + (size_t)mbase * K;
    const __bf16* Wb = W + (size_t)nbase * K;

#if HAVE_TDM
    if (wave == 0) {
        tdm_load_2d(Ab, lds_addr_of(ashm), 16, K, K);
        __builtin_amdgcn_s_wait_tensorcnt(0);
    }
#else
    for (int i = threadIdx.x * 8; i < 16 * K; i += 256 * 8)
        *(v8bf*)&ashm[i] = *(const v8bf*)&Ab[i];
#endif
    __syncthreads();

    v8f c = {};
    for (int k0 = 0; k0 < K; k0 += 32) {
        v16bf a = load_a_frag(ashm, K, k0, lane);
        v16bf b = load_b_frag(Wb, K, k0, lane);
        c = __builtin_amdgcn_wmma_f32_16x16x32_bf16(false, a, false, b, (short)0, c, false, false);
    }

    int col = nbase + (lane & 15);
    float bv = bias ? bias[col] : 0.0f;
#pragma unroll
    for (int v = 0; v < 8; ++v) {
        int row = mbase + v + ((lane >> 4) << 3);
        float val = c[v] + bv;
        size_t idx = (size_t)row * N + col;
        Cf[idx] = val;
        if (Cb) Cb[idx] = f2bf(val);
        if (Cg) Cg[idx] = f2bf(geluf_(val));
    }
}

// ---------------------------------------------------------------- K column sums
__global__ void k_ksum(const float* __restrict__ Kf, float* __restrict__ Ksum) {
    int i = blockIdx.x * blockDim.x + threadIdx.x;       // i in [0, B*DK)
    int b = i / DK_, k = i % DK_;
    const float* p = Kf + (size_t)b * S_ * DK_ + k;
    float s = 0.0f;
    for (int t = 0; t < S_; ++t) s += p[(size_t)t * DK_];
    Ksum[i] = s;
}

// ---------------------------------------------------------------- argmax of Q@K^T per query
// block = 256 (8 waves); each wave owns one 16-query tile. WMMA score tiles, cross-lane argmax.
__global__ void k_argmax(const __bf16* __restrict__ Qb, const __bf16* __restrict__ Kb,
                         int* __restrict__ maxids) {
    int lane = threadIdx.x & 31;
    int wave = threadIdx.x >> 5;
    int wid  = blockIdx.x * 8 + wave;
    int b    = wid >> 7;            // S/16 = 128
    int qt   = wid & 127;
    int qbase = qt * 16;

    const __bf16* Qp = Qb + ((size_t)b * S_ + qbase) * DK_;

    v16bf aq[16];
#pragma unroll
    for (int kc = 0; kc < 16; ++kc) aq[kc] = load_a_frag(Qp, DK_, kc * 32, lane);

    float bestv[8]; int besti[8];
#pragma unroll
    for (int v = 0; v < 8; ++v) { bestv[v] = -1e30f; besti[v] = 0; }

    for (int kb = 0; kb < S_; kb += 16) {
        const __bf16* Kp = Kb + ((size_t)b * S_ + kb) * DK_;
        v8f s = {};
#pragma unroll
        for (int kc = 0; kc < 16; ++kc) {
            v16bf bf = load_b_frag(Kp, DK_, kc * 32, lane);
            s = __builtin_amdgcn_wmma_f32_16x16x32_bf16(false, aq[kc], false, bf, (short)0, s, false, false);
        }
#pragma unroll
        for (int v = 0; v < 8; ++v) {
            float val = s[v];
            int   idx = kb + (lane & 15);
#pragma unroll
            for (int m = 1; m < 16; m <<= 1) {
                float ov = __shfl_xor(val, m, 16);
                int   oi = __shfl_xor(idx, m, 16);
                if (ov > val || (ov == val && oi < idx)) { val = ov; idx = oi; }
            }
            if (val > bestv[v]) { bestv[v] = val; besti[v] = idx; }
        }
    }
    if ((lane & 15) == 0) {
#pragma unroll
        for (int v = 0; v < 8; ++v) {
            int row = v + ((lane >> 4) << 3);
            maxids[(size_t)b * S_ + qbase + row] = besti[v];
        }
    }
}

// ---------------------------------------------------------------- gating (elementwise)
__global__ void k_gate(const float* __restrict__ Qf, const float* __restrict__ Vf,
                       const float* __restrict__ ttqRaw, const float* __restrict__ ttvRaw,
                       const float* __restrict__ Kf, const float* __restrict__ Ksum,
                       const int* __restrict__ maxids,
                       const float* __restrict__ alphaq, const float* __restrict__ alphav,
                       __bf16* __restrict__ gQ, __bf16* __restrict__ gV) {
    size_t idx = (size_t)blockIdx.x * blockDim.x + threadIdx.x;   // over B*S*DK
    int k  = (int)(idx % DK_);
    int bs = (int)(idx / DK_);
    int s  = bs % S_;
    int b  = bs / S_;

    float lp = __logf((float)(s + 1));

    float q    = Qf[idx];
    float ttq  = tanhf(ttqRaw[idx]);
    float kmax = Kf[((size_t)b * S_ + maxids[(size_t)b * S_ + s]) * DK_ + k];
    float knob = fabsf(q * (Ksum[(size_t)b * DK_ + k] - (float)S_ * kmax));
    float ptq  = 1.0f + sigmoidf_(alphaq[s] - knob) * lp;
    gQ[idx] = f2bf((ptq + ttq) * q);

    float vv  = Vf[idx];
    float ttv = tanhf(ttvRaw[idx]);
    float ptv = 1.0f + sigmoidf_(alphav[s]) * lp;
    gV[idx] = f2bf((ptv + ttv) * vv);
}

// ---------------------------------------------------------------- causal flash attention
// grid = (S/16, B), block = 256 (8 waves). Wave w owns DV cols [w*64, w*64+64).
// K and gated-V chunks (32 x 512 bf16, contiguous) staged into LDS by TDM.
__global__ void k_flash(const __bf16* __restrict__ gQ, const __bf16* __restrict__ Kb,
                        const __bf16* __restrict__ gV, float* __restrict__ out) {
    __shared__ __align__(128) __bf16 kshm[32 * DK_];        // 32 KB
    __shared__ __align__(128) __bf16 vshm[32 * DV_];        // 32 KB
    __shared__ __align__(16)  __bf16 pshm[8][16][32];       // 8 KB

    int lane = threadIdx.x & 31;
    int wave = threadIdx.x >> 5;
    int qbase = blockIdx.x * 16;
    int b     = blockIdx.y;
    int cw    = wave * 64;

    const __bf16* Qp = gQ + ((size_t)b * S_ + qbase) * DK_;
    const float scale = 0.044194173824159216f;   // 1/sqrt(512)

    float m[8], l[8];
    v8f acc[4];
#pragma unroll
    for (int v = 0; v < 8; ++v) { m[v] = -1e30f; l[v] = 0.0f; }
#pragma unroll
    for (int t = 0; t < 4; ++t) acc[t] = {};

    for (int kb = 0; kb < qbase + 16; kb += 32) {
        const __bf16* Kp = Kb + ((size_t)b * S_ + kb) * DK_;
        const __bf16* Vp = gV + ((size_t)b * S_ + kb) * DV_;

#if HAVE_TDM
        if (wave == 0) {
            tdm_load_2d(Kp, lds_addr_of(kshm), 32, DK_, DK_);
            tdm_load_2d(Vp, lds_addr_of(vshm), 32, DV_, DV_);
            __builtin_amdgcn_s_wait_tensorcnt(0);
        }
#else
        for (int i = threadIdx.x * 8; i < 32 * DK_; i += 256 * 8) {
            *(v8bf*)&kshm[i] = *(const v8bf*)&Kp[i];
            *(v8bf*)&vshm[i] = *(const v8bf*)&Vp[i];
        }
#endif
        __syncthreads();

        v8f s0 = {}, s1 = {};
#pragma unroll
        for (int kc = 0; kc < 16; ++kc) {
            v16bf a  = load_a_frag(Qp, DK_, kc * 32, lane);
            v16bf b0 = load_b_frag(kshm,             DK_, kc * 32, lane);
            v16bf b1 = load_b_frag(kshm + 16 * DK_,  DK_, kc * 32, lane);
            s0 = __builtin_amdgcn_wmma_f32_16x16x32_bf16(false, a, false, b0, (short)0, s0, false, false);
            s1 = __builtin_amdgcn_wmma_f32_16x16x32_bf16(false, a, false, b1, (short)0, s1, false, false);
        }

        int col0 = kb + (lane & 15);
        float sc[8], p0[8], p1[8];
#pragma unroll
        for (int v = 0; v < 8; ++v) {
            int rowg = qbase + v + ((lane >> 4) << 3);
            float x0 = (col0      <= rowg) ? s0[v] * scale : -1e30f;
            float x1 = (col0 + 16 <= rowg) ? s1[v] * scale : -1e30f;
            float mv = fmaxf(x0, x1);
#pragma unroll
            for (int mm = 1; mm < 16; mm <<= 1) mv = fmaxf(mv, __shfl_xor(mv, mm, 16));
            float mnew = fmaxf(m[v], mv);
            sc[v] = __expf(m[v] - mnew);
            p0[v] = __expf(x0 - mnew);
            p1[v] = __expf(x1 - mnew);
            float rs = p0[v] + p1[v];
#pragma unroll
            for (int mm = 1; mm < 16; mm <<= 1) rs += __shfl_xor(rs, mm, 16);
            l[v] = l[v] * sc[v] + rs;
            m[v] = mnew;
        }
#pragma unroll
        for (int t = 0; t < 4; ++t)
#pragma unroll
            for (int v = 0; v < 8; ++v) acc[t][v] *= sc[v];

        // transpose P (C layout -> A layout) through LDS (wave-private tile)
#pragma unroll
        for (int v = 0; v < 8; ++v) {
            int row = v + ((lane >> 4) << 3);
            pshm[wave][row][lane & 15]        = f2bf(p0[v]);
            pshm[wave][row][(lane & 15) + 16] = f2bf(p1[v]);
        }
        {
            int row = lane & 15;
            int ko  = (lane >> 4) << 4;
            v8bf lo = *(const v8bf*)&pshm[wave][row][ko];
            v8bf hi = *(const v8bf*)&pshm[wave][row][ko + 8];
            v16bf pf;
#pragma unroll
            for (int i = 0; i < 8; ++i) { pf[i] = lo[i]; pf[i + 8] = hi[i]; }

#pragma unroll
            for (int t = 0; t < 4; ++t) {
                v16bf vb = load_b_frag_kmajor(vshm, DV_, cw + t * 16, lane);
                acc[t] = __builtin_amdgcn_wmma_f32_16x16x32_bf16(false, pf, false, vb, (short)0, acc[t], false, false);
            }
        }
        __syncthreads();   // protect kshm/vshm before next chunk's TDM overwrite
    }

#pragma unroll
    for (int v = 0; v < 8; ++v) {
        int row = v + ((lane >> 4) << 3);
        float linv = 1.0f / l[v];
        size_t base = ((size_t)b * S_ + qbase + row) * DV_ + cw + (lane & 15);
#pragma unroll
        for (int t = 0; t < 4; ++t) out[base + t * 16] = acc[t][v] * linv;
    }
}

// ---------------------------------------------------------------- host
extern "C" void kernel_launch(void* const* d_in, const int* in_sizes, int n_in,
                              void* d_out, int out_size, void* d_ws, size_t ws_size,
                              hipStream_t stream) {
    (void)in_sizes; (void)n_in; (void)out_size; (void)ws_size;
    const float* x      = (const float*)d_in[0];
    const float* Wq_w   = (const float*)d_in[1];
    const float* Wq_b   = (const float*)d_in[2];
    const float* Wk_w   = (const float*)d_in[3];
    const float* Wk_b   = (const float*)d_in[4];
    const float* Wv_w   = (const float*)d_in[5];
    const float* Wv_b   = (const float*)d_in[6];
    const float* alphaq = (const float*)d_in[7];
    const float* alphav = (const float*)d_in[8];
    const float* tokenq = (const float*)d_in[9];
    const float* tokenv = (const float*)d_in[10];
    float* out = (float*)d_out;

    const int M = B_ * S_;
    char* ws = (char*)d_ws;
    size_t off = 0;
    auto alloc = [&](size_t bytes) -> char* {
        char* p = ws + off;
        off = (off + bytes + 255) & ~(size_t)255;
        return p;
    };

    __bf16* xb   = (__bf16*)alloc((size_t)M * E_ * 2);
    __bf16* Wqb  = (__bf16*)alloc((size_t)DK_ * E_ * 2);
    __bf16* Wkb  = (__bf16*)alloc((size_t)DK_ * E_ * 2);
    __bf16* Wvb  = (__bf16*)alloc((size_t)DV_ * E_ * 2);
    __bf16* tqb  = (__bf16*)alloc((size_t)DK_ * DK_ * 2);
    __bf16* tvb  = (__bf16*)alloc((size_t)DV_ * DV_ * 2);
    float*  Qf   = (float*) alloc((size_t)M * DK_ * 4);
    float*  Kf   = (float*) alloc((size_t)M * DK_ * 4);
    float*  Vf   = (float*) alloc((size_t)M * DV_ * 4);
    __bf16* Qb   = (__bf16*)alloc((size_t)M * DK_ * 2);
    __bf16* Kb   = (__bf16*)alloc((size_t)M * DK_ * 2);
    __bf16* Qg   = (__bf16*)alloc((size_t)M * DK_ * 2);
    __bf16* Vg   = (__bf16*)alloc((size_t)M * DV_ * 2);
    float*  ttq  = (float*) alloc((size_t)M * DK_ * 4);
    float*  ttv  = (float*) alloc((size_t)M * DV_ * 4);
    __bf16* gQ   = (__bf16*)alloc((size_t)M * DK_ * 2);
    __bf16* gV   = (__bf16*)alloc((size_t)M * DV_ * 2);
    float*  Ksum = (float*) alloc((size_t)B_ * DK_ * 4);
    int*    mids = (int*)   alloc((size_t)B_ * S_ * 4);

    auto cvt = [&](const float* s, __bf16* d, int n) {
        k_f32_to_bf16<<<(n + 255) / 256, 256, 0, stream>>>(s, d, n);
    };
    cvt(x, xb, M * E_);
    cvt(Wq_w, Wqb, DK_ * E_);
    cvt(Wk_w, Wkb, DK_ * E_);
    cvt(Wv_w, Wvb, DV_ * E_);
    cvt(tokenq, tqb, DK_ * DK_);
    cvt(tokenv, tvb, DV_ * DV_);

    dim3 gproj(DK_ / 128, M / 16);
    k_gemm_bf16<<<gproj, 256, 0, stream>>>(xb, Wqb, Wq_b, Qf, Qb, Qg, M, DK_, E_);
    k_gemm_bf16<<<gproj, 256, 0, stream>>>(xb, Wkb, Wk_b, Kf, Kb, nullptr, M, DK_, E_);
    k_gemm_bf16<<<gproj, 256, 0, stream>>>(xb, Wvb, Wv_b, Vf, nullptr, Vg, M, DV_, E_);
    k_gemm_bf16<<<gproj, 256, 0, stream>>>(Qg, tqb, nullptr, ttq, nullptr, nullptr, M, DK_, DK_);
    k_gemm_bf16<<<gproj, 256, 0, stream>>>(Vg, tvb, nullptr, ttv, nullptr, nullptr, M, DV_, DV_);

    k_ksum<<<(B_ * DK_) / 256, 256, 0, stream>>>(Kf, Ksum);
    k_argmax<<<(B_ * (S_ / 16)) / 8, 256, 0, stream>>>(Qb, Kb, mids);
    k_gate<<<(int)(((size_t)M * DK_) / 256), 256, 0, stream>>>(Qf, Vf, ttq, ttv, Kf, Ksum, mids,
                                                               alphaq, alphav, gQ, gV);
    k_flash<<<dim3(S_ / 16, B_), 256, 0, stream>>>(gQ, Kb, gV, out);
}
